// TransformerBlock_6038724018983
// MI455X (gfx1250) — matrix operand order repaired and checked by reading, not compiled, not run on hardware
//
#include <hip/hip_runtime.h>
#include <hip/hip_bf16.h>

#define HD     768
#define NHEAD  12
#define DKDIM  64
#define BATCH  4
#define SLEN   2048
#define MROWS  (BATCH * SLEN)
#define FFDIM  3072

typedef __attribute__((ext_vector_type(16))) __bf16 bf16x16;
typedef __attribute__((ext_vector_type(8)))  __bf16 bf16x8;
typedef __attribute__((ext_vector_type(8)))  float  f32x8;

union FragU { bf16x16 v; bf16x8 h[2]; };

// A fragment: 16x32 (MxK) bf16, rows striped across lanes (M = lane%16),
// per lane two contiguous 8-element runs at k0 + half*8 and k0 + half*8 + 16.
static __device__ __forceinline__ bf16x16
load_frag_a(const __bf16* base, int ld, int row0, int k0, int lane) {
  int m  = row0 + (lane & 15);
  int hf = lane >> 4;
  const __bf16* p = base + (size_t)m * ld + k0 + hf * 8;
  FragU f;
  f.h[0] = *(const bf16x8*)(p);
  f.h[1] = *(const bf16x8*)(p + 16);
  return f.v;
}

// B fragment: 32x16 (KxN) bf16, loaded from a row-major N x K (i.e. B^T)
// matrix: column N = lane%16, per lane two contiguous 8-element runs at
// k0 + half*8 and k0 + half*8 + 16.
static __device__ __forceinline__ bf16x16
load_frag_b(const __bf16* baseT, int ld, int col0, int k0, int lane) {
  int n  = col0 + (lane & 15);
  int hf = lane >> 4;
  const __bf16* p = baseT + (size_t)n * ld + k0 + hf * 8;
  FragU f;
  f.h[0] = *(const bf16x8*)(p);
  f.h[1] = *(const bf16x8*)(p + 16);
  return f.v;
}

static __device__ __forceinline__ f32x8
wmma_bf16(bf16x16 a, bf16x16 b, f32x8 c) {
  return __builtin_amdgcn_wmma_f32_16x16x32_bf16(
      /*neg_a=*/false, a, /*neg_b=*/false, b,
      /*c_mod=*/(short)0, c, /*reuse_a=*/false, /*reuse_b=*/false);
}

// -------------------------------------------------------------------------
// Weight convert + transpose: w[K][N] f32 -> wT[N][K] bf16
// -------------------------------------------------------------------------
__global__ void wconv_kernel(const float* __restrict__ w, __bf16* __restrict__ wT,
                             int K, int N) {
  int idx = blockIdx.x * blockDim.x + threadIdx.x;
  if (idx >= K * N) return;
  int n = idx / K;
  int k = idx % K;
  wT[idx] = (__bf16)w[(size_t)k * N + n];
}

// -------------------------------------------------------------------------
// LayerNorm: one 256-thread block per row (768 elems, 3 per thread), bf16 out
// -------------------------------------------------------------------------
__global__ __launch_bounds__(256)
void ln_kernel(const float* __restrict__ x, const float* __restrict__ w,
               const float* __restrict__ bvec, __bf16* __restrict__ y) {
  __shared__ float red[256];
  int row = blockIdx.x;
  int t = threadIdx.x;
  const float* xr = x + (size_t)row * HD;
  float v0 = xr[t], v1 = xr[t + 256], v2 = xr[t + 512];
  red[t] = v0 + v1 + v2;
  __syncthreads();
  for (int o = 128; o > 0; o >>= 1) { if (t < o) red[t] += red[t + o]; __syncthreads(); }
  float mean = red[0] * (1.0f / HD);
  __syncthreads();
  float d0 = v0 - mean, d1 = v1 - mean, d2 = v2 - mean;
  red[t] = d0 * d0 + d1 * d1 + d2 * d2;
  __syncthreads();
  for (int o = 128; o > 0; o >>= 1) { if (t < o) red[t] += red[t + o]; __syncthreads(); }
  float inv = rsqrtf(red[0] * (1.0f / HD) + 1e-12f);
  __bf16* yr = y + (size_t)row * HD;
  yr[t]       = (__bf16)(w[t]       * (d0 * inv) + bvec[t]);
  yr[t + 256] = (__bf16)(w[t + 256] * (d1 * inv) + bvec[t + 256]);
  yr[t + 512] = (__bf16)(w[t + 512] * (d2 * inv) + bvec[t + 512]);
}

// -------------------------------------------------------------------------
// bf16 WMMA GEMM: C[M,N] = A[M,K] @ B[K,N], B given as B^T (N x K).
// 128 threads = 4 waves; each wave owns a 64x64 tile (4x4 WMMA tiles,
// 16 WMMAs per 32-deep k-step, 32 FLOP/byte of fragment traffic),
// block tile 128x128 so A/B slices are shared in WGP$ across the 4 waves.
// MODE 0: store bf16 C[M,N]
// MODE 1: store bf16 to V^T layout [B,NH,DK,S]
// MODE 2: store f32  C + bias + residual
// MODE 3: store bf16 gelu(C + bias)
// -------------------------------------------------------------------------
template <int MODE>
__global__ __launch_bounds__(128)
void gemm_kernel(const __bf16* __restrict__ A, const __bf16* __restrict__ BT,
                 const float* __restrict__ bias, const float* __restrict__ resid,
                 void* __restrict__ out, int Mdim, int Ndim, int Kdim) {
  const int lane = threadIdx.x & 31;
  const int wave = threadIdx.x >> 5;
  const int waveM = blockIdx.x * 128 + (wave >> 1) * 64;
  const int waveN = blockIdx.y * 128 + (wave & 1) * 64;

  const f32x8 zero = {0.f, 0.f, 0.f, 0.f, 0.f, 0.f, 0.f, 0.f};
  f32x8 acc[4][4];
  #pragma unroll
  for (int i = 0; i < 4; ++i)
    #pragma unroll
    for (int j = 0; j < 4; ++j) acc[i][j] = zero;

  for (int k0 = 0; k0 < Kdim; k0 += 32) {
    bf16x16 a[4], b[4];
    #pragma unroll
    for (int i = 0; i < 4; ++i) a[i] = load_frag_a(A, Kdim, waveM + i * 16, k0, lane);
    #pragma unroll
    for (int j = 0; j < 4; ++j) b[j] = load_frag_b(BT, Kdim, waveN + j * 16, k0, lane);
    if (k0 + 32 < Kdim) {  // pull next k-slice toward the caches (global_prefetch_b8)
      __builtin_prefetch(A  + (size_t)(waveM + (lane & 15)) * Kdim + k0 + 32, 0, 1);
      __builtin_prefetch(A  + (size_t)(waveM + 32 + (lane & 15)) * Kdim + k0 + 32, 0, 1);
      __builtin_prefetch(BT + (size_t)(waveN + (lane & 15)) * Kdim + k0 + 32, 0, 1);
      __builtin_prefetch(BT + (size_t)(waveN + 32 + (lane & 15)) * Kdim + k0 + 32, 0, 1);
    }
    #pragma unroll
    for (int i = 0; i < 4; ++i)
      #pragma unroll
      for (int j = 0; j < 4; ++j)
        acc[i][j] = wmma_bf16(a[i], b[j], acc[i][j]);
  }

  const int hf = lane >> 4;
  const int nl = lane & 15;
  #pragma unroll
  for (int i = 0; i < 4; ++i) {
    #pragma unroll
    for (int j = 0; j < 4; ++j) {
      int tm = waveM + i * 16 + hf * 8;
      int tn = waveN + j * 16 + nl;
      #pragma unroll
      for (int r = 0; r < 8; ++r) {
        int m = tm + r;
        float v = acc[i][j][r];
        if (MODE == 0) {
          ((__bf16*)out)[(size_t)m * Ndim + tn] = (__bf16)v;
        } else if (MODE == 1) {
          int b2 = m / SLEN, s = m % SLEN;
          int h = tn / DKDIM, d = tn % DKDIM;
          ((__bf16*)out)[((size_t)(b2 * NHEAD + h) * DKDIM + d) * SLEN + s] = (__bf16)v;
        } else if (MODE == 2) {
          ((float*)out)[(size_t)m * Ndim + tn] =
              v + bias[tn] + resid[(size_t)m * Ndim + tn];
        } else {
          float u = v + bias[tn];
          float g = 0.5f * u *
              (1.0f + tanhf(0.7978845608028654f * (u + 0.044715f * u * u * u)));
          ((__bf16*)out)[(size_t)m * Ndim + tn] = (__bf16)g;
        }
      }
    }
  }
}

// -------------------------------------------------------------------------
// Flash attention: one wave per (16-row q tile, head, batch). Streams keys in
// chunks of 32 with online softmax; P re-laid to A-fragment via 1KB LDS.
// -------------------------------------------------------------------------
__global__ __launch_bounds__(32)
void attn_kernel(const __bf16* __restrict__ Q, const __bf16* __restrict__ Kmat,
                 const __bf16* __restrict__ VT, const int* __restrict__ mask,
                 __bf16* __restrict__ ctx) {
  __shared__ __bf16 pLds[16 * 32];
  const int lane = threadIdx.x;
  const int q0 = blockIdx.x * 16;
  const int h  = blockIdx.y;
  const int b  = blockIdx.z;
  const int hf = lane >> 4;
  const int nl = lane & 15;
  const int rowBase = b * SLEN;

  bf16x16 qa0 = load_frag_a(Q, HD, rowBase + q0, h * DKDIM,      lane);
  bf16x16 qa1 = load_frag_a(Q, HD, rowBase + q0, h * DKDIM + 32, lane);
  const __bf16* vbase = VT + (size_t)(b * NHEAD + h) * DKDIM * SLEN;

  int mq[8];
  #pragma unroll
  for (int r = 0; r < 8; ++r) mq[r] = mask[rowBase + q0 + hf * 8 + r];

  const f32x8 zero = {0.f, 0.f, 0.f, 0.f, 0.f, 0.f, 0.f, 0.f};
  f32x8 acc0 = zero, acc1 = zero, acc2 = zero, acc3 = zero;
  float mrun[8], lrun[8];
  #pragma unroll
  for (int r = 0; r < 8; ++r) { mrun[r] = -3.0e38f; lrun[r] = 0.0f; }

  for (int kc = 0; kc < SLEN; kc += 32) {
    bf16x16 kb00 = load_frag_b(Kmat, HD, rowBase + kc,      h * DKDIM,      lane);
    bf16x16 kb01 = load_frag_b(Kmat, HD, rowBase + kc,      h * DKDIM + 32, lane);
    bf16x16 kb10 = load_frag_b(Kmat, HD, rowBase + kc + 16, h * DKDIM,      lane);
    bf16x16 kb11 = load_frag_b(Kmat, HD, rowBase + kc + 16, h * DKDIM + 32, lane);

    f32x8 s0 = zero, s1 = zero;
    s0 = wmma_bf16(qa0, kb00, s0);
    s0 = wmma_bf16(qa1, kb01, s0);
    s1 = wmma_bf16(qa0, kb10, s1);
    s1 = wmma_bf16(qa1, kb11, s1);

    int mk0 = mask[rowBase + kc + nl];
    int mk1 = mask[rowBase + kc + 16 + nl];

    #pragma unroll
    for (int r = 0; r < 8; ++r) {
      float a = s0[r] * 0.125f;                 // 1/sqrt(64)
      float c = s1[r] * 0.125f;
      if (!mk0 || !mq[r]) a = -1.0e9f;
      if (!mk1 || !mq[r]) c = -1.0e9f;
      // rowwise max over the 32-key chunk (16-lane half holds one row)
      float t = fmaxf(a, c);
      t = fmaxf(t, __shfl_xor(t, 1, 32));
      t = fmaxf(t, __shfl_xor(t, 2, 32));
      t = fmaxf(t, __shfl_xor(t, 4, 32));
      t = fmaxf(t, __shfl_xor(t, 8, 32));
      float mnew  = fmaxf(mrun[r], t);
      float alpha = __expf(mrun[r] - mnew);
      mrun[r] = mnew;
      float p0 = __expf(a - mnew);
      float p1 = __expf(c - mnew);
      float rs = p0 + p1;
      rs += __shfl_xor(rs, 1, 32);
      rs += __shfl_xor(rs, 2, 32);
      rs += __shfl_xor(rs, 4, 32);
      rs += __shfl_xor(rs, 8, 32);
      lrun[r] = lrun[r] * alpha + rs;
      acc0[r] *= alpha; acc1[r] *= alpha; acc2[r] *= alpha; acc3[r] *= alpha;
      int prow = hf * 8 + r;
      pLds[prow * 32 + nl]      = (__bf16)p0;
      pLds[prow * 32 + 16 + nl] = (__bf16)p1;
    }
    __syncthreads();

    bf16x16 pa = load_frag_a(pLds, 32, 0, 0, lane);
    bf16x16 v0 = load_frag_b(vbase, SLEN, 0,  kc, lane);
    bf16x16 v1 = load_frag_b(vbase, SLEN, 16, kc, lane);
    bf16x16 v2 = load_frag_b(vbase, SLEN, 32, kc, lane);
    bf16x16 v3 = load_frag_b(vbase, SLEN, 48, kc, lane);
    acc0 = wmma_bf16(pa, v0, acc0);
    acc1 = wmma_bf16(pa, v1, acc1);
    acc2 = wmma_bf16(pa, v2, acc2);
    acc3 = wmma_bf16(pa, v3, acc3);
    __syncthreads();
  }

  #pragma unroll
  for (int r = 0; r < 8; ++r) {
    float inv = 1.0f / lrun[r];
    size_t orow = (size_t)(rowBase + q0 + hf * 8 + r) * HD + h * DKDIM;
    ctx[orow + 0  + nl] = (__bf16)(acc0[r] * inv);
    ctx[orow + 16 + nl] = (__bf16)(acc1[r] * inv);
    ctx[orow + 32 + nl] = (__bf16)(acc2[r] * inv);
    ctx[orow + 48 + nl] = (__bf16)(acc3[r] * inv);
  }
}

// -------------------------------------------------------------------------
extern "C" void kernel_launch(void* const* d_in, const int* in_sizes, int n_in,
                              void* d_out, int out_size, void* d_ws, size_t ws_size,
                              hipStream_t stream) {
  const float* x     = (const float*)d_in[0];
  const int*   mask  = (const int*)d_in[1];
  const float* wq    = (const float*)d_in[2];
  const float* wk    = (const float*)d_in[3];
  const float* wv    = (const float*)d_in[4];
  const float* wo    = (const float*)d_in[5];
  const float* wo_b  = (const float*)d_in[6];
  const float* w1    = (const float*)d_in[7];
  const float* w1_b  = (const float*)d_in[8];
  const float* w2    = (const float*)d_in[9];
  const float* w2_b  = (const float*)d_in[10];
  const float* ln1_w = (const float*)d_in[11];
  const float* ln1_b = (const float*)d_in[12];
  const float* ln2_w = (const float*)d_in[13];
  const float* ln2_b = (const float*)d_in[14];
  (void)in_sizes; (void)n_in; (void)out_size; (void)ws_size;

  char* ws = (char*)d_ws;
  size_t off = 0;
  auto carve = [&](size_t bytes) -> char* {
    char* p = ws + off;
    off += (bytes + 255) & ~(size_t)255;
    return p;
  };

  __bf16* wqT  = (__bf16*)carve((size_t)HD * HD * 2);
  __bf16* wkT  = (__bf16*)carve((size_t)HD * HD * 2);
  __bf16* wvT  = (__bf16*)carve((size_t)HD * HD * 2);
  __bf16* woT  = (__bf16*)carve((size_t)HD * HD * 2);
  __bf16* w1T  = (__bf16*)carve((size_t)HD * FFDIM * 2);
  __bf16* w2T  = (__bf16*)carve((size_t)HD * FFDIM * 2);
  __bf16* ybuf = (__bf16*)carve((size_t)MROWS * HD * 2);      // y1, reused as y2
  char*   big  = carve((size_t)MROWS * FFDIM * 2);            // q,k,vT,ctx; reused as h
  __bf16* qbuf = (__bf16*)big;
  __bf16* kbuf = (__bf16*)(big + (size_t)MROWS * HD * 2);
  __bf16* vT   = (__bf16*)(big + (size_t)MROWS * HD * 4);
  __bf16* ctx  = (__bf16*)(big + (size_t)MROWS * HD * 6);
  __bf16* hbuf = (__bf16*)big;
  float*  x1   = (float*)carve((size_t)MROWS * HD * 4);

  const int nHH = HD * HD, nHF = HD * FFDIM;
  wconv_kernel<<<(nHH + 255) / 256, 256, 0, stream>>>(wq, wqT, HD, HD);
  wconv_kernel<<<(nHH + 255) / 256, 256, 0, stream>>>(wk, wkT, HD, HD);
  wconv_kernel<<<(nHH + 255) / 256, 256, 0, stream>>>(wv, wvT, HD, HD);
  wconv_kernel<<<(nHH + 255) / 256, 256, 0, stream>>>(wo, woT, HD, HD);
  wconv_kernel<<<(nHF + 255) / 256, 256, 0, stream>>>(w1, w1T, HD, FFDIM);
  wconv_kernel<<<(nHF + 255) / 256, 256, 0, stream>>>(w2, w2T, FFDIM, HD);

  ln_kernel<<<MROWS, 256, 0, stream>>>(x, ln1_w, ln1_b, ybuf);

  dim3 gH(MROWS / 128, HD / 128), blk(128);
  gemm_kernel<0><<<gH, blk, 0, stream>>>(ybuf, wqT, nullptr, nullptr, qbuf, MROWS, HD, HD);
  gemm_kernel<0><<<gH, blk, 0, stream>>>(ybuf, wkT, nullptr, nullptr, kbuf, MROWS, HD, HD);
  gemm_kernel<1><<<gH, blk, 0, stream>>>(ybuf, wvT, nullptr, nullptr, vT,   MROWS, HD, HD);

  attn_kernel<<<dim3(SLEN / 16, NHEAD, BATCH), 32, 0, stream>>>(qbuf, kbuf, vT, mask, ctx);

  gemm_kernel<2><<<gH, blk, 0, stream>>>(ctx, woT, wo_b, x, x1, MROWS, HD, HD);

  ln_kernel<<<MROWS, 256, 0, stream>>>(x1, ln2_w, ln2_b, ybuf);

  gemm_kernel<3><<<dim3(MROWS / 128, FFDIM / 128), blk, 0, stream>>>(
      ybuf, w1T, w1_b, nullptr, hbuf, MROWS, FFDIM, HD);
  gemm_kernel<2><<<gH, blk, 0, stream>>>(
      hbuf, w2T, w2_b, x1, (float*)d_out, MROWS, HD, FFDIM);
}
